// CaptioningRNN_69767448756333
// MI455X (gfx1250) — matrix operand order repaired and checked by reading, not compiled, not run on hardware
//
#include <hip/hip_runtime.h>
#include <math.h>

// ---------------- problem constants ----------------
#define NB   128          // batch N
#define TT   128          // timesteps
#define DD   1024         // input dim
#define HH   1024         // hidden dim
#define KK   3072         // D + 2H  (concat GEMM reduction dim)
#define FH   4096         // 4*H

typedef __attribute__((ext_vector_type(16))) __bf16 v16bf;
typedef __attribute__((ext_vector_type(8)))  __bf16 v8bf;
typedef __attribute__((ext_vector_type(8)))  float  v8f;

__device__ __forceinline__ unsigned short f2bf_rne(float f) {
    union { float f; unsigned int u; } v; v.f = f;
    unsigned int u = v.u;
    unsigned int r = u + 0x7FFFu + ((u >> 16) & 1u);   // round to nearest even
    return (unsigned short)(r >> 16);
}

__device__ __forceinline__ float sigm(float x) {
    return 1.0f / (1.0f + __expf(-x));
}

// ---------------------------------------------------------------------------
// Kernel 1: pack [Wx; Wh; Wattn] into column-major bf16  WcatT[col][k]
// so a WMMA B-fragment is 16 contiguous K-values per lane (one 32B load).
// ---------------------------------------------------------------------------
__global__ void convert_weights_kernel(const float* __restrict__ Wx,
                                       const float* __restrict__ Wh,
                                       const float* __restrict__ Wa,
                                       unsigned short* __restrict__ WcatT) {
    int idx = blockIdx.x * blockDim.x + threadIdx.x;   // idx = col*KK + k
    int col = idx / KK;
    int k   = idx - col * KK;
    float v;
    if (k < DD)            v = Wx[k * FH + col];
    else if (k < DD + HH)  v = Wh[(k - DD) * FH + col];
    else                   v = Wa[(k - DD - HH) * FH + col];
    WcatT[idx] = f2bf_rne(v);
}

// ---------------------------------------------------------------------------
// Kernel 2: h0 = c0 = meanpool_{4x4}(A)
// ---------------------------------------------------------------------------
__global__ void init_state_kernel(const float* __restrict__ A,
                                  float* __restrict__ h,
                                  float* __restrict__ c) {
    int idx = blockIdx.x * blockDim.x + threadIdx.x;   // n*HH + j
    const float* Ar = A + (size_t)idx * 16;
    float s = 0.f;
    #pragma unroll
    for (int k = 0; k < 16; ++k) s += Ar[k];
    s *= (1.0f / 16.0f);
    h[idx] = s;
    c[idx] = s;
}

// ---------------------------------------------------------------------------
// Kernel 3 (per step): 16-cell attention softmax over A_flat with current h,
// then write bf16 Acat row n = [ x_t | h | attn ]   (128 x 3072)
// One workgroup per batch row n.
// ---------------------------------------------------------------------------
__global__ void attn_pack_kernel(const float* __restrict__ x,
                                 const float* __restrict__ A,
                                 const float* __restrict__ h,
                                 unsigned short* __restrict__ Acat,
                                 int t) {
    __shared__ float red[16 * 256];
    __shared__ float wsh[16];
    const int n   = blockIdx.x;
    const int tid = threadIdx.x;
    const float scale = 0.03125f;                       // 1/sqrt(1024)

    // partial logits: logit[k] = sum_h h[n][h]*A[n][h][k]
    float p[16];
    #pragma unroll
    for (int k = 0; k < 16; ++k) p[k] = 0.f;
    for (int hh = tid; hh < HH; hh += 256) {
        float hv = h[n * HH + hh];
        const float* Ar = A + ((size_t)(n * HH + hh)) * 16;
        #pragma unroll
        for (int k = 0; k < 16; ++k) p[k] += hv * Ar[k];
    }
    #pragma unroll
    for (int k = 0; k < 16; ++k) red[k * 256 + tid] = p[k];
    __syncthreads();
    for (int s = 128; s > 0; s >>= 1) {
        if (tid < s) {
            #pragma unroll
            for (int k = 0; k < 16; ++k)
                red[k * 256 + tid] += red[k * 256 + tid + s];
        }
        __syncthreads();
    }
    if (tid == 0) {                                     // serial 16-wide softmax
        float m = -3.0e38f;
        #pragma unroll
        for (int k = 0; k < 16; ++k) {
            float v = red[k * 256] * scale;
            m = fmaxf(m, v);
        }
        float sum = 0.f;
        float e[16];
        #pragma unroll
        for (int k = 0; k < 16; ++k) {
            e[k] = __expf(red[k * 256] * scale - m);
            sum += e[k];
        }
        float inv = 1.0f / sum;
        #pragma unroll
        for (int k = 0; k < 16; ++k) wsh[k] = e[k] * inv;
    }
    __syncthreads();

    // attn[n][hh] = sum_k A[n][hh][k]*w[k];  pack Acat = [x_t | h | attn] bf16
    unsigned short* arow = Acat + (size_t)n * KK;
    for (int hh = tid; hh < HH; hh += 256) {
        const float* Ar = A + ((size_t)(n * HH + hh)) * 16;
        float av = 0.f;
        #pragma unroll
        for (int k = 0; k < 16; ++k) av += Ar[k] * wsh[k];
        arow[hh]            = f2bf_rne(x[((size_t)(n * TT + t)) * DD + hh]);
        arow[DD + hh]       = f2bf_rne(h[n * HH + hh]);
        arow[DD + HH + hh]  = f2bf_rne(av);
    }
}

// ---------------------------------------------------------------------------
// Kernel 4 (per step): a = Acat(128x3072) @ WcatT^T(3072x4096) + b, fused gates.
// Each wave computes a 32-row x 16-col tile for ALL FOUR gates (cols j,
// j+1024, j+2048, j+3072): 2 A-fragments x 4 B-fragments -> 8 chained
// v_wmma_f32_16x16x32_bf16 per 32-K step.  This halves the L2 weight stream
// (each weight element is read once per 32-row tile -> 4x per step instead of
// 8x) and gives 8 independent WMMA accumulation chains per wave for latency
// hiding.  4 row-tiles x 64 col-tiles = 256 wave tiles = 64 WGs x 4 waves.
// LSTM gate update happens register-resident in the epilogue.
// ---------------------------------------------------------------------------
__global__ void lstm_step_kernel(const unsigned short* __restrict__ Acat,
                                 const unsigned short* __restrict__ WcatT,
                                 const float* __restrict__ bias,
                                 float* __restrict__ h,
                                 float* __restrict__ c,
                                 float* __restrict__ out,
                                 int t) {
    const int wave = threadIdx.x >> 5;
    const int lane = threadIdx.x & 31;
    const int l15  = lane & 15;
    const int kh   = lane >> 4;            // half-wave select
    const int kh8  = kh * 16;              // A-frag K sub-offset
    const int kb16 = kh * 16;              // B-frag K sub-offset

    const int tile    = blockIdx.x * 4 + wave;   // 0..255
    const int rowBase = (tile >> 6) << 5;        // 4 row tiles of 32 (batch)
    const int colBase = (tile & 63) << 4;        // 64 col tiles of 16 (hidden)

    // A fragment sources: two 16-row blocks of Acat (row-major, K contiguous)
    const unsigned short* arow0 = Acat + (size_t)(rowBase + l15) * KK + kh8;
    const unsigned short* arow1 = Acat + (size_t)(rowBase + 16 + l15) * KK + kh8;
    // B fragment sources: one column of WcatT per gate (K contiguous)
    const unsigned short* bcol0 = WcatT + (size_t)(0 * HH + colBase + l15) * KK + kb16;
    const unsigned short* bcol1 = WcatT + (size_t)(1 * HH + colBase + l15) * KK + kb16;
    const unsigned short* bcol2 = WcatT + (size_t)(2 * HH + colBase + l15) * KK + kb16;
    const unsigned short* bcol3 = WcatT + (size_t)(3 * HH + colBase + l15) * KK + kb16;

    v8f acc00 = {}, acc01 = {}, acc02 = {}, acc03 = {};   // rows [0,16)
    v8f acc10 = {}, acc11 = {}, acc12 = {}, acc13 = {};   // rows [16,32)

    #pragma unroll 2
    for (int k0 = 0; k0 < KK; k0 += 32) {
        // A-fragments: lane(kh) holds 16 contiguous K in [kh*16, kh*16+16)
        // for its row (matches B).
        v8bf lo0 = *(const v8bf*)(arow0 + k0);
        v8bf hi0 = *(const v8bf*)(arow0 + k0 + 8);
        v8bf lo1 = *(const v8bf*)(arow1 + k0);
        v8bf hi1 = *(const v8bf*)(arow1 + k0 + 8);
        v16bf a0, a1;
        #pragma unroll
        for (int i = 0; i < 8; ++i) {
            a0[i] = lo0[i]; a0[i + 8] = hi0[i];
            a1[i] = lo1[i]; a1[i + 8] = hi1[i];
        }

        // B-fragments: 16 contiguous K values for this lane's column.
        v16bf b0 = *(const v16bf*)(bcol0 + k0);
        v16bf b1 = *(const v16bf*)(bcol1 + k0);
        v16bf b2 = *(const v16bf*)(bcol2 + k0);
        v16bf b3 = *(const v16bf*)(bcol3 + k0);

        acc00 = __builtin_amdgcn_wmma_f32_16x16x32_bf16(false, a0, false, b0,
                                                        (short)0, acc00, false, false);
        acc10 = __builtin_amdgcn_wmma_f32_16x16x32_bf16(false, a1, false, b0,
                                                        (short)0, acc10, false, false);
        acc01 = __builtin_amdgcn_wmma_f32_16x16x32_bf16(false, a0, false, b1,
                                                        (short)0, acc01, false, false);
        acc11 = __builtin_amdgcn_wmma_f32_16x16x32_bf16(false, a1, false, b1,
                                                        (short)0, acc11, false, false);
        acc02 = __builtin_amdgcn_wmma_f32_16x16x32_bf16(false, a0, false, b2,
                                                        (short)0, acc02, false, false);
        acc12 = __builtin_amdgcn_wmma_f32_16x16x32_bf16(false, a1, false, b2,
                                                        (short)0, acc12, false, false);
        acc03 = __builtin_amdgcn_wmma_f32_16x16x32_bf16(false, a0, false, b3,
                                                        (short)0, acc03, false, false);
        acc13 = __builtin_amdgcn_wmma_f32_16x16x32_bf16(false, a1, false, b3,
                                                        (short)0, acc13, false, false);
    }

    // Epilogue: bias + LSTM gates, register-resident.
    // D layout: column fixed per lane (colBase+l15); rows = base + kh*8 + r.
    const int col = colBase + l15;
    const float bi = bias[0 * HH + col];
    const float bf = bias[1 * HH + col];
    const float bo = bias[2 * HH + col];
    const float bg = bias[3 * HH + col];

    #pragma unroll
    for (int m = 0; m < 2; ++m) {
        const v8f* a_i = m ? &acc10 : &acc00;
        const v8f* a_f = m ? &acc11 : &acc01;
        const v8f* a_o = m ? &acc12 : &acc02;
        const v8f* a_g = m ? &acc13 : &acc03;
        #pragma unroll
        for (int r = 0; r < 8; ++r) {
            int row = rowBase + m * 16 + kh * 8 + r;   // batch index
            int idx = row * HH + col;
            float ai = (*a_i)[r] + bi;
            float af = (*a_f)[r] + bf;
            float ao = (*a_o)[r] + bo;
            float ag = (*a_g)[r] + bg;
            float cold = c[idx];
            float cn = sigm(af) * cold + sigm(ai) * tanhf(ag);
            float hn = sigm(ao) * tanhf(cn);
            c[idx] = cn;
            h[idx] = hn;
            out[((size_t)row * TT + t) * HH + col] = hn;
        }
    }
}

// ---------------------------------------------------------------------------
// Host launcher
// ---------------------------------------------------------------------------
extern "C" void kernel_launch(void* const* d_in, const int* in_sizes, int n_in,
                              void* d_out, int out_size, void* d_ws, size_t ws_size,
                              hipStream_t stream) {
    (void)in_sizes; (void)n_in; (void)out_size; (void)ws_size;

    const float* x     = (const float*)d_in[0];   // (N,T,D)
    const float* A     = (const float*)d_in[1];   // (N,H,4,4) == (N,H,16)
    const float* Wx    = (const float*)d_in[2];   // (D,4H)
    const float* Wh    = (const float*)d_in[3];   // (H,4H)
    const float* Wattn = (const float*)d_in[4];   // (H,4H)
    const float* bvec  = (const float*)d_in[5];   // (4H)
    float* out = (float*)d_out;                   // (N,T,H)

    // workspace layout
    char* base = (char*)d_ws;
    const size_t WCAT_BYTES = (size_t)FH * KK * sizeof(unsigned short); // 24 MB
    const size_t ACAT_BYTES = (size_t)NB * KK * sizeof(unsigned short); // 768 KB
    const size_t H_BYTES    = (size_t)NB * HH * sizeof(float);          // 512 KB
    unsigned short* WcatT = (unsigned short*)(base);
    unsigned short* Acat  = (unsigned short*)(base + WCAT_BYTES);
    float* hbuf = (float*)(base + WCAT_BYTES + ACAT_BYTES);
    float* cbuf = (float*)(base + WCAT_BYTES + ACAT_BYTES + H_BYTES);

    // 1) bf16 weight pack (col-major, stacked [Wx;Wh;Wattn])
    {
        int total  = FH * KK;                 // 12,582,912
        int blocks = total / 256;
        convert_weights_kernel<<<blocks, 256, 0, stream>>>(Wx, Wh, Wattn, WcatT);
    }
    // 2) h0 = c0 = meanpool(A)
    {
        int total  = NB * HH;                 // 131,072
        int blocks = total / 256;
        init_state_kernel<<<blocks, 256, 0, stream>>>(A, hbuf, cbuf);
    }
    // 3) sequential recurrence: attention/pack, then fused WMMA GEMM + gates
    for (int t = 0; t < TT; ++t) {
        attn_pack_kernel<<<NB, 256, 0, stream>>>(x, A, hbuf, Acat, t);
        lstm_step_kernel<<<64, 128, 0, stream>>>(Acat, WcatT, bvec,
                                                 hbuf, cbuf, out, t);
    }
}
